// sLSTM_54571854463592
// MI455X (gfx1250) — matrix operand order repaired and checked by reading, not compiled, not run on hardware
//
#include <hip/hip_runtime.h>
#include <hip/hip_bf16.h>
#include <stdint.h>

typedef __attribute__((ext_vector_type(16))) __bf16 v16bf;
typedef __attribute__((ext_vector_type(8)))  float  v8f;
typedef __attribute__((ext_vector_type(4)))  unsigned int u32x4;
typedef __attribute__((ext_vector_type(8)))  int i32x8;
typedef __attribute__((ext_vector_type(4)))  int i32x4;
typedef __bf16 bf16;

#define B_SZ   4096
#define F_SZ   2048
#define HD     2048
#define PROJ_N 2730
#define UPN    5460
#define KP     2752          // padded PROJ (86*32)
#define LN_EPS_C 2048.0f     // float(F) per reference
#define GN_EPS_C 1e-6f
#define EPS_C    1e-8f
#define AP 40                // LDS pitch (elements) -> conflict-free fragment gathers

#define USE_TDM 1            // Tensor Data Mover for A-tile staging (6-arg clang-23 builtin)

// ---------------- fragment helpers (CDNA5 16-bit WMMA layouts) ----------------
union FragU  { uint4 q[2]; v16bf v; };
union B16x8  { uint4 q;    bf16 h[8];  };
union B16x4  { uint2 q;    bf16 h[4];  };
union B16x2  { unsigned q; bf16 h[2];  };

// A 16x32: lane m (0..15), kh=lane>>4; elems 0..7 -> K=kh*8+e ; 8..15 -> K=16+kh*8+(e-8)
__device__ __forceinline__ v16bf frag_a(const bf16* sA, int m, int kh) {
    FragU u;
    const bf16* p = sA + m * AP + kh * 8;
    u.q[0] = *(const uint4*)p;
    u.q[1] = *(const uint4*)(p + 16);
    return u.v;
}
// B 32x16 (staged transposed in LDS as [n][k]): lane n, kh; elems 0..7 -> K=kh*8+e ; 8..15 -> K=16+kh*8+(e-8)
__device__ __forceinline__ v16bf frag_b(const bf16* sBt, int n, int kh) {
    FragU u;
    const bf16* p = sBt + n * AP + kh * 8;
    u.q[0] = *(const uint4*)p;
    u.q[1] = *(const uint4*)(p + 16);
    return u.v;
}
__device__ __forceinline__ v8f wmma_bf16(v16bf a, v16bf b, v8f c) {
    return __builtin_amdgcn_wmma_f32_16x16x32_bf16(false, a, false, b, (short)0, c, false, false);
}

// ---------------- TDM: DMA one 64-row x 32-col bf16 tile into pitch-40 LDS ----------------
// D# per CDNA5 ISA 8.3/8.4: data_size=2B, tile_dim0=32 (K), tile_dim1=64 (rows),
// pad_enable, pad_interval=16 DWORDs (row = 64B), pad_amount=4 DWORDs (16B) -> 80B pitch.
__device__ __forceinline__ void tdm_load_tile64x32(unsigned lds_off,
                                                   unsigned long long gaddr,
                                                   unsigned ld_elems) {
#if USE_TDM
    u32x4 g0;
    g0.x = 1u;                                           // count=1 (valid user D#)
    g0.y = lds_off;                                      // lds_addr
    g0.z = (unsigned)(gaddr & 0xffffffffull);            // global_addr[31:0]
    g0.w = (unsigned)((gaddr >> 32) & 0x01ffffffull)     // global_addr[56:32]
         | (2u << 30);                                   // type=2 ("image")
    i32x8 g1;
    g1[0] = (int)((1u << 16)                             // data_size = 2 bytes
                | (1u << 20)                             // pad_enable
                | (3u << 22)                             // pad_interval: 16 DWORDs
                | (3u << 25));                           // pad_amount: 4 DWORDs
    g1[1] = (int)((ld_elems & 0xffffu) << 16);           // tensor_dim0[15:0]
    g1[2] = (int)(((ld_elems >> 16) & 0xffffu)           // tensor_dim0[31:16]
                | ((unsigned)B_SZ & 0xffffu) << 16);     // tensor_dim1[15:0]
    g1[3] = (int)((32u & 0xffffu) << 16);                // tile_dim0 = 32 cols
    g1[4] = (int)(64u);                                  // tile_dim1 = 64 rows
    g1[5] = (int)ld_elems;                               // tensor_dim0_stride[31:0]
    g1[6] = 0;                                           // stride high bits
    g1[7] = 0;
    i32x4 z4 = {0, 0, 0, 0};
    i32x8 z8 = {0, 0, 0, 0, 0, 0, 0, 0};
    __builtin_amdgcn_tensor_load_to_lds(g0, g1, z4, z4, z8, 0);
#endif
}
__device__ __forceinline__ void tdm_wait() {
#if USE_TDM
    __builtin_amdgcn_s_wait_tensorcnt(0);
#endif
}
__device__ __forceinline__ unsigned lds_off_of(const void* p) {
    return (unsigned)(uintptr_t)p;                       // flat LDS addr truncates to LDS offset
}

// ---------------- math helpers ----------------
__device__ __forceinline__ float sigmoidf_(float x) { return 1.0f / (1.0f + __expf(-x)); }
__device__ __forceinline__ float log_sigmoidf_(float x) {
    return fminf(x, 0.0f) - log1pf(__expf(-fabsf(x)));
}
__device__ __forceinline__ float gelu_tanhf_(float x) {
    float x3 = x * x * x;
    return 0.5f * x * (1.0f + tanhf(0.7978845608028654f * (x + 0.044715f * x3)));
}

// ================= LayerNorm + window shift + bf16 cast =================
__global__ __launch_bounds__(256) void ln_shift_kernel(
    const float* __restrict__ inputs, const float* __restrict__ x_prev,
    const float* __restrict__ lns, const float* __restrict__ lnb,
    bf16* __restrict__ xt, float* __restrict__ xprev_new) {
    __shared__ float ssum[256], ssq[256];
    int b = blockIdx.x, t = threadIdx.x;
    const float* row = inputs + (size_t)b * F_SZ;
    int j0 = t * 8;
    float v[8];
    float4 a0 = *(const float4*)(row + j0);
    float4 a1 = *(const float4*)(row + j0 + 4);
    v[0]=a0.x; v[1]=a0.y; v[2]=a0.z; v[3]=a0.w;
    v[4]=a1.x; v[5]=a1.y; v[6]=a1.z; v[7]=a1.w;
    float s = 0.f, q = 0.f;
#pragma unroll
    for (int e = 0; e < 8; ++e) { s += v[e]; q += v[e] * v[e]; }
    ssum[t] = s; ssq[t] = q; __syncthreads();
    for (int off = 128; off > 0; off >>= 1) {
        if (t < off) { ssum[t] += ssum[t + off]; ssq[t] += ssq[t + off]; }
        __syncthreads();
    }
    float mu  = ssum[0] * (1.0f / F_SZ);
    float var = ssq[0] * (1.0f / F_SZ) - mu * mu;
    float rin = rsqrtf(var + LN_EPS_C);
    float* dst2 = xprev_new + ((size_t)b * 3 + 2) * F_SZ;
    bf16* xrow = xt + (size_t)b * F_SZ;
#pragma unroll
    for (int e = 0; e < 8; ++e) {
        int j = j0 + e;
        float y = (v[e] - mu) * rin * lns[j] + lnb[j];
        dst2[j] = y;
        xrow[j] = (bf16)y;
    }
    const float* xp = x_prev + (size_t)b * 3 * F_SZ;
    float* dst0 = xprev_new + (size_t)b * 3 * F_SZ;
#pragma unroll
    for (int e = 0; e < 8; ++e) {
        dst0[j0 + e]        = xp[F_SZ + j0 + e];
        dst0[F_SZ + j0 + e] = xp[2 * F_SZ + j0 + e];
    }
}

// ================= per-row all(n==0) flags =================
__global__ __launch_bounds__(256) void flags_kernel(const float* __restrict__ n,
                                                    int* __restrict__ flags) {
    __shared__ int sa[256];
    int b = blockIdx.x, t = threadIdx.x;
    const float* row = n + (size_t)b * HD;
    int any = 0;
#pragma unroll
    for (int e = 0; e < 8; ++e) any |= (row[t * 8 + e] != 0.0f);
    sa[t] = any; __syncthreads();
    for (int off = 128; off > 0; off >>= 1) {
        if (t < off) sa[t] |= sa[t + off];
        __syncthreads();
    }
    if (t == 0) flags[b] = (sa[0] == 0) ? 1 : 0;
}

// ================= f32 -> bf16 converters =================
__global__ __launch_bounds__(256) void cvt_bf16_kernel(const float* __restrict__ src,
                                                       bf16* __restrict__ dst, int count) {
    int i = blockIdx.x * 256 + threadIdx.x;
    int stride = gridDim.x * 256;
    for (; i < count; i += stride) dst[i] = (bf16)src[i];
}
__global__ __launch_bounds__(256) void cvt_downk_kernel(const float* __restrict__ src,
                                                        bf16* __restrict__ dst) {
    int total = KP * F_SZ;
    int i = blockIdx.x * 256 + threadIdx.x;
    int stride = gridDim.x * 256;
    for (; i < total; i += stride) {
        int r = i >> 11, c = i & (F_SZ - 1);
        dst[i] = (r < PROJ_N) ? (bf16)src[r * F_SZ + c] : (bf16)0.0f;
    }
}

// ================= fused 4-gate GEMM + sLSTM state update =================
struct GParams {
    const bf16* xt;                  // [B, HD] bf16
    const bf16* w;                   // [4][HD][HD] bf16 (z,i,f,o)
    const float* wb[4];
    const float* rk[4];              // [16][8][8]
    const float* rb[4];              // [16][8]
    const float* c; const float* n; const float* h; const float* m;
    const int* flags;
    float* ct; float* nt; float* ht; float* mt;
};

__global__ __launch_bounds__(256) void gates_kernel(GParams P) {
    __shared__ bf16 sA[2][64 * AP];
    __shared__ bf16 sB[2][4][32 * AP];
    __shared__ float sR[4][1024];
    __shared__ float sRb[4][128];

    int t = threadIdx.x;
    int lane = t & 31, wv = t >> 5;
    int lane15 = lane & 15, kh = lane >> 4;
    int ml = (wv >> 1) * 16, nl = (wv & 1) * 16;
    int m0 = blockIdx.x * 64, n0 = blockIdx.y * 32;

#pragma unroll
    for (int g = 0; g < 4; ++g) {
        for (int i = t; i < 1024; i += 256) sR[g][i] = P.rk[g][i];
        for (int i = t; i < 128;  i += 256) sRb[g][i] = P.rb[g][i];
    }

    v8f zero = {0.f,0.f,0.f,0.f,0.f,0.f,0.f,0.f};
    v8f acc[4] = {zero, zero, zero, zero};

    // B staging: 4 gate tiles of 32x32, 16 elems/thread = 4 rows x 4 cols, b64 packed stores
    int g4 = t >> 6, idx = t & 63;
    int krb = (idx >> 3) * 4, c0 = (idx & 7) * 4;

    unsigned long long gaA = (unsigned long long)(uintptr_t)P.xt + (size_t)m0 * HD * 2;
    const bf16* wsrc0 = P.w + ((size_t)g4 * HD + krb) * HD + n0 + c0;

    B16x4 brow[4];
    // ---- prologue: tile 0 ----
    if (wv == 0) tdm_load_tile64x32(lds_off_of(&sA[0][0]), gaA, HD);
#if !USE_TDM
    {   int arow = t >> 2, aoff = (t & 3) * 8;
        *(uint4*)&sA[0][arow * AP + aoff] = *(const uint4*)&P.xt[(size_t)(m0 + arow) * HD + aoff]; }
#endif
#pragma unroll
    for (int r = 0; r < 4; ++r) brow[r].q = *(const uint2*)(wsrc0 + (size_t)r * HD);
#pragma unroll
    for (int cc = 0; cc < 4; ++cc) {
        B16x4 u;
#pragma unroll
        for (int r = 0; r < 4; ++r) u.h[r] = brow[r].h[cc];
        *(uint2*)&sB[0][g4][(c0 + cc) * AP + krb] = u.q;
    }
    if (wv == 0) tdm_wait();
    __syncthreads();

    for (int k0 = 0; k0 < HD; k0 += 32) {
        int cur = (k0 >> 5) & 1, nxt = cur ^ 1;
        bool more = (k0 + 32) < HD;
        if (more) {
            if (wv == 0)
                tdm_load_tile64x32(lds_off_of(&sA[nxt][0]), gaA + (size_t)(k0 + 32) * 2, HD);
            const bf16* wsrc = wsrc0 + (size_t)(k0 + 32) * HD;
#pragma unroll
            for (int r = 0; r < 4; ++r) brow[r].q = *(const uint2*)(wsrc + (size_t)r * HD);
        }
        // ---- compute: group all fragment loads, then back-to-back WMMAs ----
        v16bf af = frag_a(sA[cur], ml + lane15, kh);
        v16bf b0 = frag_b(sB[cur][0], nl + lane15, kh);
        v16bf b1 = frag_b(sB[cur][1], nl + lane15, kh);
        v16bf b2 = frag_b(sB[cur][2], nl + lane15, kh);
        v16bf b3 = frag_b(sB[cur][3], nl + lane15, kh);
        acc[0] = wmma_bf16(af, b0, acc[0]);
        acc[1] = wmma_bf16(af, b1, acc[1]);
        acc[2] = wmma_bf16(af, b2, acc[2]);
        acc[3] = wmma_bf16(af, b3, acc[3]);
        if (more) {
#if !USE_TDM
            {   int arow = t >> 2, aoff = (t & 3) * 8;
                *(uint4*)&sA[nxt][arow * AP + aoff] =
                    *(const uint4*)&P.xt[(size_t)(m0 + arow) * HD + k0 + 32 + aoff]; }
#endif
#pragma unroll
            for (int cc = 0; cc < 4; ++cc) {
                B16x4 u;
#pragma unroll
                for (int r = 0; r < 4; ++r) u.h[r] = brow[r].h[cc];
                *(uint2*)&sB[nxt][g4][(c0 + cc) * AP + krb] = u.q;
            }
            if (wv == 0) tdm_wait();
        }
        __syncthreads();
    }

    // ---- epilogue: recurrence + sLSTM pointwise ----
    int j = n0 + nl + lane15;
    int nb = (j >> 3) & 15, e = j & 7;
    float Rc[4][8], bias[4];
#pragma unroll
    for (int g = 0; g < 4; ++g) {
#pragma unroll
        for (int d = 0; d < 8; ++d) Rc[g][d] = sR[g][nb * 64 + d * 8 + e];
        bias[g] = P.wb[g][j] + sRb[g][nb * 8 + e];
    }
#pragma unroll
    for (int v = 0; v < 8; ++v) {
        int bi = m0 + ml + v + 8 * kh;
        size_t ix = (size_t)bi * HD + j;
        const float* hp = P.h + (size_t)bi * HD + (j & ~7);
        float4 h0 = *(const float4*)hp;
        float4 h1 = *(const float4*)(hp + 4);
        float hv[8] = {h0.x, h0.y, h0.z, h0.w, h1.x, h1.y, h1.z, h1.w};
        float rec[4] = {0.f, 0.f, 0.f, 0.f};
#pragma unroll
        for (int d = 0; d < 8; ++d) {
#pragma unroll
            for (int g = 0; g < 4; ++g) rec[g] += hv[d] * Rc[g][d];
        }
        float z_raw = acc[0][v] + bias[0] + rec[0];
        float i_raw = acc[1][v] + bias[1] + rec[1];
        float f_raw = acc[2][v] + bias[2] + rec[2];
        float o_raw = acc[3][v] + bias[3] + rec[3];

        float mm = P.m[ix];
        float logfplusm = mm + log_sigmoidf_(f_raw);
        float m_t = P.flags[bi] ? i_raw : fmaxf(i_raw, logfplusm);
        float o_t = sigmoidf_(o_raw);
        float i_t = __expf(i_raw - m_t);
        float f_t = __expf(logfplusm - m_t);
        float z_t = tanhf(z_raw);
        float cv = P.c[ix], nv = P.n[ix];
        float c_t = f_t * cv + i_t * z_t;
        float n_t = f_t * nv + i_t;
        float h_t = o_t * (c_t / (n_t + EPS_C));
        P.ct[ix] = c_t; P.nt[ix] = n_t; P.ht[ix] = h_t; P.mt[ix] = m_t;
    }
}

// ================= GroupNorm (groups across heads) =================
__global__ __launch_bounds__(256) void gn_kernel(const float* __restrict__ ht,
                                                 const float* __restrict__ gns,
                                                 const float* __restrict__ gnb,
                                                 bf16* __restrict__ outn) {
    __shared__ float ssum[256], ssq[256], gmu[16], grs[16];
    int b = blockIdx.x, t = threadIdx.x;
    const float* row = ht + (size_t)b * HD;
    float v[8];
    float4 a0 = *(const float4*)(row + t * 8);
    float4 a1 = *(const float4*)(row + t * 8 + 4);
    v[0]=a0.x; v[1]=a0.y; v[2]=a0.z; v[3]=a0.w;
    v[4]=a1.x; v[5]=a1.y; v[6]=a1.z; v[7]=a1.w;
    float s = 0.f, q = 0.f;
#pragma unroll
    for (int e = 0; e < 8; ++e) { s += v[e]; q += v[e] * v[e]; }
    ssum[t] = s; ssq[t] = q; __syncthreads();
    if (t < 16) {
        float S = 0.f, Q = 0.f;
        for (int i = t; i < 256; i += 16) { S += ssum[i]; Q += ssq[i]; }
        float mu = S * (1.0f / 128.0f);
        float var = Q * (1.0f / 128.0f) - mu * mu;
        gmu[t] = mu; grs[t] = rsqrtf(var + GN_EPS_C);
    }
    __syncthreads();
    int g = t & 15;
    float mu = gmu[g], rs = grs[g];
    bf16* orow = outn + (size_t)b * HD + t * 8;
#pragma unroll
    for (int e = 0; e < 8; ++e) {
        int dcol = g * 8 + e;
        orow[e] = (bf16)((v[e] - mu) * rs * gns[dcol] + gnb[dcol]);
    }
}

// ================= up GEMM (o1 & o2 column pair) + GLU/gelu =================
__global__ __launch_bounds__(256) void up_kernel(const bf16* __restrict__ outn,
                                                 const bf16* __restrict__ upk,
                                                 const float* __restrict__ upb,
                                                 bf16* __restrict__ act) {
    __shared__ bf16 sA[2][64 * AP];
    __shared__ bf16 sB[2][2][32 * AP];
    int t = threadIdx.x;
    int lane = t & 31, wv = t >> 5;
    int lane15 = lane & 15, kh = lane >> 4;
    int ml = (wv >> 1) * 16, nl = (wv & 1) * 16;
    int m0 = blockIdx.x * 64, n0 = blockIdx.y * 32;

    v8f acc1 = {0.f,0.f,0.f,0.f,0.f,0.f,0.f,0.f};
    v8f acc2 = acc1;

    // B staging: 2 tiles, 8 elems/thread = 2 rows x 4 cols (guarded scalar loads, b32 stores)
    int sel = t >> 7, idx = t & 127;
    int krb = (idx >> 3) * 2, c0 = (idx & 7) * 4;
    int cbase = sel ? PROJ_N : 0;

    unsigned long long gaA = (unsigned long long)(uintptr_t)outn + (size_t)m0 * HD * 2;
    bf16 bres[2][4];

    auto loadB = [&](int k0) {
#pragma unroll
        for (int r = 0; r < 2; ++r) {
            const bf16* src = upk + (size_t)(k0 + krb + r) * UPN + cbase;
#pragma unroll
            for (int cc = 0; cc < 4; ++cc) {
                int col = n0 + c0 + cc;
                bres[r][cc] = (col < PROJ_N) ? src[col] : (bf16)0.0f;
            }
        }
    };
    auto storeB = [&](int buf) {
#pragma unroll
        for (int cc = 0; cc < 4; ++cc) {
            B16x2 u; u.h[0] = bres[0][cc]; u.h[1] = bres[1][cc];
            *(unsigned*)&sB[buf][sel][(c0 + cc) * AP + krb] = u.q;
        }
    };

    if (wv == 0) tdm_load_tile64x32(lds_off_of(&sA[0][0]), gaA, HD);
#if !USE_TDM
    {   int arow = t >> 2, aoff = (t & 3) * 8;
        *(uint4*)&sA[0][arow * AP + aoff] = *(const uint4*)&outn[(size_t)(m0 + arow) * HD + aoff]; }
#endif
    loadB(0); storeB(0);
    if (wv == 0) tdm_wait();
    __syncthreads();

    for (int k0 = 0; k0 < HD; k0 += 32) {
        int cur = (k0 >> 5) & 1, nxt = cur ^ 1;
        bool more = (k0 + 32) < HD;
        if (more) {
            if (wv == 0)
                tdm_load_tile64x32(lds_off_of(&sA[nxt][0]), gaA + (size_t)(k0 + 32) * 2, HD);
            loadB(k0 + 32);
        }
        v16bf af = frag_a(sA[cur], ml + lane15, kh);
        v16bf b1 = frag_b(sB[cur][0], nl + lane15, kh);
        v16bf b2 = frag_b(sB[cur][1], nl + lane15, kh);
        acc1 = wmma_bf16(af, b1, acc1);
        acc2 = wmma_bf16(af, b2, acc2);
        if (more) {
#if !USE_TDM
            {   int arow = t >> 2, aoff = (t & 3) * 8;
                *(uint4*)&sA[nxt][arow * AP + aoff] =
                    *(const uint4*)&outn[(size_t)(m0 + arow) * HD + k0 + 32 + aoff]; }
#endif
            storeB(nxt);
            if (wv == 0) tdm_wait();
        }
        __syncthreads();
    }

    int col = n0 + nl + lane15;
    float ub1 = (col < PROJ_N) ? upb[col] : 0.0f;
    float ub2 = (col < PROJ_N) ? upb[col + PROJ_N] : 0.0f;
#pragma unroll
    for (int v = 0; v < 8; ++v) {
        int bi = m0 + ml + v + 8 * kh;
        float a = 0.0f;
        if (col < PROJ_N) {
            float u1 = acc1[v] + ub1;
            float u2 = acc2[v] + ub2;
            a = u1 * gelu_tanhf_(u2);
        }
        act[(size_t)bi * KP + col] = (bf16)a;
    }
}

// ================= down GEMM + bias + residual =================
__global__ __launch_bounds__(256) void down_kernel(const bf16* __restrict__ act,
                                                   const bf16* __restrict__ dk,
                                                   const float* __restrict__ db,
                                                   const float* __restrict__ inputs,
                                                   float* __restrict__ out) {
    __shared__ bf16 sA[2][64 * AP];
    __shared__ bf16 sB[2][32 * AP];
    int t = threadIdx.x;
    int lane = t & 31, wv = t >> 5;
    int lane15 = lane & 15, kh = lane >> 4;
    int ml = (wv >> 1) * 16, nl = (wv & 1) * 16;
    int m0 = blockIdx.x * 64, n0 = blockIdx.y * 32;

    v8f acc = {0.f,0.f,0.f,0.f,0.f,0.f,0.f,0.f};

    // B staging: 4 elems/thread = 2 rows x 2 cols, uint loads, b32 packed stores
    int krb = (t >> 4) * 2, c0 = (t & 15) * 2;
    unsigned long long gaA = (unsigned long long)(uintptr_t)act + (size_t)m0 * KP * 2;
    B16x2 brow[2];

    auto loadB = [&](int k0) {
#pragma unroll
        for (int r = 0; r < 2; ++r)
            brow[r].q = *(const unsigned*)(dk + (size_t)(k0 + krb + r) * F_SZ + n0 + c0);
    };
    auto storeB = [&](int buf) {
#pragma unroll
        for (int cc = 0; cc < 2; ++cc) {
            B16x2 u; u.h[0] = brow[0].h[cc]; u.h[1] = brow[1].h[cc];
            *(unsigned*)&sB[buf][(c0 + cc) * AP + krb] = u.q;
        }
    };

    if (wv == 0) tdm_load_tile64x32(lds_off_of(&sA[0][0]), gaA, KP);
#if !USE_TDM
    {   int arow = t >> 2, aoff = (t & 3) * 8;
        *(uint4*)&sA[0][arow * AP + aoff] = *(const uint4*)&act[(size_t)(m0 + arow) * KP + aoff]; }
#endif
    loadB(0); storeB(0);
    if (wv == 0) tdm_wait();
    __syncthreads();

    for (int k0 = 0; k0 < KP; k0 += 32) {
        int cur = (k0 >> 5) & 1, nxt = cur ^ 1;
        bool more = (k0 + 32) < KP;
        if (more) {
            if (wv == 0)
                tdm_load_tile64x32(lds_off_of(&sA[nxt][0]), gaA + (size_t)(k0 + 32) * 2, KP);
            loadB(k0 + 32);
        }
        v16bf af = frag_a(sA[cur], ml + lane15, kh);
        v16bf bfr = frag_b(sB[cur], nl + lane15, kh);
        acc = wmma_bf16(af, bfr, acc);
        if (more) {
#if !USE_TDM
            {   int arow = t >> 2, aoff = (t & 3) * 8;
                *(uint4*)&sA[nxt][arow * AP + aoff] =
                    *(const uint4*)&act[(size_t)(m0 + arow) * KP + k0 + 32 + aoff]; }
#endif
            storeB(nxt);
            if (wv == 0) tdm_wait();
        }
        __syncthreads();
    }

    int col = n0 + nl + lane15;
    float bias = db[col];
#pragma unroll
    for (int v = 0; v < 8; ++v) {
        int bi = m0 + ml + v + 8 * kh;
        size_t ix = (size_t)bi * F_SZ + col;
        out[ix] = acc[v] + bias + inputs[ix];
    }
}

// =======================================================================
extern "C" void kernel_launch(void* const* d_in, const int* in_sizes, int n_in,
                              void* d_out, int out_size, void* d_ws, size_t ws_size,
                              hipStream_t stream) {
    const float* inputs = (const float*)d_in[0];
    const float* c      = (const float*)d_in[1];
    const float* n      = (const float*)d_in[2];
    const float* h      = (const float*)d_in[3];
    const float* m      = (const float*)d_in[4];
    const float* x_prev = (const float*)d_in[5];
    const float* ln_s   = (const float*)d_in[6];
    const float* ln_b   = (const float*)d_in[7];
    const float* Wk[4]  = {(const float*)d_in[8],  (const float*)d_in[12],
                           (const float*)d_in[16], (const float*)d_in[20]};
    const float* Wb[4]  = {(const float*)d_in[9],  (const float*)d_in[13],
                           (const float*)d_in[17], (const float*)d_in[21]};
    const float* Rk[4]  = {(const float*)d_in[10], (const float*)d_in[14],
                           (const float*)d_in[18], (const float*)d_in[22]};
    const float* Rb[4]  = {(const float*)d_in[11], (const float*)d_in[15],
                           (const float*)d_in[19], (const float*)d_in[23]};
    const float* gn_s   = (const float*)d_in[24];
    const float* gn_b   = (const float*)d_in[25];
    const float* up_k   = (const float*)d_in[26];
    const float* up_b   = (const float*)d_in[27];
    const float* down_k = (const float*)d_in[28];
    const float* down_b = (const float*)d_in[29];

    float* out = (float*)d_out;
    const size_t E = (size_t)B_SZ * HD;
    float* o_out = out;
    float* o_ct  = out + E;
    float* o_nt  = out + 2 * E;
    float* o_ht  = out + 3 * E;
    float* o_mt  = out + 4 * E;
    float* o_xp  = out + 5 * E;

    char* ws = (char*)d_ws;
    size_t off = 0;
    bf16* xt_bf  = (bf16*)(ws + off); off += (size_t)B_SZ * F_SZ * 2;
    bf16* w_bf   = (bf16*)(ws + off); off += (size_t)4 * HD * HD * 2;
    bf16* upk_bf = (bf16*)(ws + off); off += (size_t)HD * UPN * 2;
    bf16* dk_bf  = (bf16*)(ws + off); off += (size_t)KP * F_SZ * 2;
    bf16* outn   = (bf16*)(ws + off); off += (size_t)B_SZ * HD * 2;
    bf16* act    = (bf16*)(ws + off); off += (size_t)B_SZ * KP * 2;
    int*  flags  = (int*)(ws + off);  off += (size_t)B_SZ * 4;

    ln_shift_kernel<<<dim3(B_SZ), dim3(256), 0, stream>>>(inputs, x_prev, ln_s, ln_b,
                                                          xt_bf, o_xp);
    flags_kernel<<<dim3(B_SZ), dim3(256), 0, stream>>>(n, flags);
    for (int g = 0; g < 4; ++g)
        cvt_bf16_kernel<<<dim3(4096), dim3(256), 0, stream>>>(
            Wk[g], w_bf + (size_t)g * HD * HD, HD * HD);
    cvt_bf16_kernel<<<dim3(4096), dim3(256), 0, stream>>>(up_k, upk_bf, HD * UPN);
    cvt_downk_kernel<<<dim3(4096), dim3(256), 0, stream>>>(down_k, dk_bf);

    GParams P;
    P.xt = xt_bf; P.w = w_bf;
    for (int g = 0; g < 4; ++g) { P.wb[g] = Wb[g]; P.rk[g] = Rk[g]; P.rb[g] = Rb[g]; }
    P.c = c; P.n = n; P.h = h; P.m = m; P.flags = flags;
    P.ct = o_ct; P.nt = o_nt; P.ht = o_ht; P.mt = o_mt;
    gates_kernel<<<dim3(64, 64), dim3(256), 0, stream>>>(P);

    gn_kernel<<<dim3(B_SZ), dim3(256), 0, stream>>>(o_ht, gn_s, gn_b, outn);
    up_kernel<<<dim3(64, 86), dim3(256), 0, stream>>>(outn, upk_bf, up_b, act);
    down_kernel<<<dim3(64, 64), dim3(256), 0, stream>>>(act, dk_bf, down_b, inputs, o_out);
}
